// PAM_Module_14955076125496
// MI455X (gfx1250) — matrix operand order repaired and checked by reading, not compiled, not run on hardware
//
#include <hip/hip_runtime.h>

// PAM dual-stream attention for MI455X (gfx1250, wave32, WMMA).
// B=4, C=256, C8=32, N=4096.
//   K1: q/k projections (fp32 VALU, 0.5 GFLOP) -> bf16 packs in WMMA A/B fragment layout
//   K2: v projections via v_wmma_f32_16x16x32_bf16 -> vhT/vlT bf16 packs (B-frag layout)
//   K3: softmax row stats (online max/sum) via WMMA energy tiles
//   K4: fused energy-recompute + P + P@V (WMMA) + gamma-gating, coalesced f32 stores
// Workspace: qPack 1MB | kPack 1MB | vhPack 8MB | vlPack 8MB | rowmax/rowsum 128KB (~18.2MB, L2-resident)

#define BB  4
#define CC  256
#define C8  32
#define NN  4096
#define NT  256   // NN/16

typedef __attribute__((ext_vector_type(16))) __bf16        v16bf;
typedef __attribute__((ext_vector_type(8)))  float         v8f;
typedef __attribute__((ext_vector_type(8)))  unsigned int  v8u;

__device__ __forceinline__ v16bf load_frag(const __bf16* p) {
    return __builtin_bit_cast(v16bf, *reinterpret_cast<const v8u*>(p));
}
__device__ __forceinline__ v8f wmma_bf16(v16bf a, v16bf b, v8f c) {
    return __builtin_amdgcn_wmma_f32_16x16x32_bf16(false, a, false, b, (short)0, c, false, false);
}
__device__ __forceinline__ v8f vzero8() {
    v8f z;
#pragma unroll
    for (int i = 0; i < 8; ++i) z[i] = 0.0f;
    return z;
}

// ---------------- K1: q/k projections, packed into WMMA fragment layouts ----------------
// qPack[b][nt][lane][elem]: A-frag (16n x 32k bf16):
//   lane = (n&15) + 16*((o>>3)&1), elem = (o&7) + 8*(o>=16)
// kPack[b][mt][lane][elem]: B-frag (32k x 16m bf16):
//   lane = (m&15) + 16*((o>>3)&1), elem = (o&7) + 8*(o>=16)
__global__ __launch_bounds__(256) void k_proj_qk(
    const float* __restrict__ x, const float* __restrict__ Wq, const float* __restrict__ bq,
    const float* __restrict__ Wk, const float* __restrict__ bk,
    __bf16* __restrict__ qPack, __bf16* __restrict__ kPack)
{
    int idx = blockIdx.x * 256 + threadIdx.x;
    int n = idx & (NN - 1);
    int o = (idx >> 12) & 31;
    int b = idx >> 17;
    const float* xcol = x + (size_t)b * CC * NN + n;
    const float* wq = Wq + o * CC;
    const float* wk = Wk + o * CC;
    float aq = bq[o], ak = bk[o];
#pragma unroll 8
    for (int c = 0; c < CC; ++c) {
        float xv = xcol[(size_t)c * NN];
        aq = fmaf(wq[c], xv, aq);
        ak = fmaf(wk[c], xv, ak);
    }
    size_t tb = ((size_t)b * NT + (n >> 4)) * 32;
    int laneQ = (n & 15) + (((o >> 3) & 1) << 4);
    int elemQ = (o & 7) + ((o >= 16) ? 8 : 0);
    qPack[((tb + laneQ) << 4) + elemQ] = (__bf16)aq;
    int laneK = (n & 15) + (((o >> 3) & 1) << 4);
    kPack[((tb + laneK) << 4) + elemQ] = (__bf16)ak;
}

// ---------------- K2: value projections via WMMA ----------------
// vT[m][c] = sum_k Wv[c,k]*x[k,m] + bv[c];  D tile rows=m, cols=c.
// Stored into B-frag pack for K4: tile = 32(K=m) x 16(c):
//   vPack[b][m/32][c/16][lane][elem], lane = (c&15)+16*((m>>4)&1), elem = m&15
__global__ __launch_bounds__(256) void k_proj_v(
    const float* __restrict__ xh1, const float* __restrict__ xl1,
    const float* __restrict__ Wv, const float* __restrict__ bv,
    __bf16* __restrict__ vhPack, __bf16* __restrict__ vlPack)
{
    int lane = threadIdx.x & 31;
    int wg = blockIdx.x * 8 + (threadIdx.x >> 5);
    int cb = wg & 3;            // 64-channel block
    int mt = (wg >> 2) & (NT - 1);
    int b  = wg >> 10;
    int m0 = mt << 4;
    int m  = m0 + (lane & 15);
    int hi = lane >> 4;
    int kbA = hi * 8;           // A-frag K base per lane-half
    int kbB = hi * 16;          // unused (B-frag load also uses kbA)

    v8f acch[4], accl[4];
#pragma unroll
    for (int i = 0; i < 4; ++i) { acch[i] = vzero8(); accl[i] = vzero8(); }

    for (int kc = 0; kc < 8; ++kc) {
        int k0 = kc * 32;
        v16bf ah, al;
#pragma unroll
        for (int e = 0; e < 16; ++e) {
            int K = k0 + kbA + (e & 7) + ((e >= 8) ? 16 : 0);
            size_t xi = (size_t)(b * CC + K) * NN + m;
            ah[e] = (__bf16)xh1[xi];
            al[e] = (__bf16)xl1[xi];
        }
#pragma unroll
        for (int ct = 0; ct < 4; ++ct) {
            int c = cb * 64 + ct * 16 + (lane & 15);
            v16bf wv;
#pragma unroll
            for (int e = 0; e < 16; ++e)
                wv[e] = (__bf16)Wv[(size_t)c * CC + k0 + kbA + (e & 7) + ((e >= 8) ? 16 : 0)];
            acch[ct] = wmma_bf16(ah, wv, acch[ct]);
            accl[ct] = wmma_bf16(al, wv, accl[ct]);
        }
    }
#pragma unroll
    for (int ct = 0; ct < 4; ++ct) {
        int c = cb * 64 + ct * 16 + (lane & 15);
        float bias = bv[c];
        size_t base = (((size_t)b * (NN / 32) + (mt >> 1)) * (CC / 16) + (cb * 4 + ct)) * 512
                      + (size_t)((lane & 15) + ((mt & 1) << 4)) * 16;
#pragma unroll
        for (int r = 0; r < 8; ++r) {
            int ml = r + (hi << 3);   // m & 15
            vhPack[base + ml] = (__bf16)(acch[ct][r] + bias);
            vlPack[base + ml] = (__bf16)(accl[ct][r] + bias);
        }
    }
}

// ---------------- K3: softmax row statistics (online max / sum-exp) ----------------
__global__ __launch_bounds__(256) void k_stats(
    const __bf16* __restrict__ qPack, const __bf16* __restrict__ kPack,
    float* __restrict__ rowmax, float* __restrict__ rowsum)
{
    int lane = threadIdx.x & 31;
    int wg = blockIdx.x * 8 + (threadIdx.x >> 5);
    int nt = wg & (NT - 1);
    int b  = wg >> 8;
    v16bf aq = load_frag(qPack + ((((size_t)b * NT + nt) * 32 + lane) << 4));
    float rmax[8], rsum[8];
#pragma unroll
    for (int r = 0; r < 8; ++r) { rmax[r] = -3.0e38f; rsum[r] = 0.0f; }
    for (int mt = 0; mt < NT; ++mt) {
        v16bf kf = load_frag(kPack + ((((size_t)b * NT + mt) * 32 + lane) << 4));
        v8f e = wmma_bf16(aq, kf, vzero8());
#pragma unroll
        for (int r = 0; r < 8; ++r) {
            float ev = e[r];
            float nm = fmaxf(rmax[r], ev);
            rsum[r] = rsum[r] * __expf(rmax[r] - nm) + __expf(ev - nm);
            rmax[r] = nm;
        }
    }
    // reduce across the 16 lanes of each half (rows 0-7 in lanes 0-15, rows 8-15 in 16-31)
#pragma unroll
    for (int r = 0; r < 8; ++r) {
#pragma unroll
        for (int mask = 1; mask < 16; mask <<= 1) {
            float om = __shfl_xor(rmax[r], mask, 32);
            float os = __shfl_xor(rsum[r], mask, 32);
            float nm = fmaxf(rmax[r], om);
            rsum[r] = rsum[r] * __expf(rmax[r] - nm) + os * __expf(om - nm);
            rmax[r] = nm;
        }
    }
    if ((lane & 15) == 0) {
#pragma unroll
        for (int r = 0; r < 8; ++r) {
            int n = nt * 16 + r + ((lane >> 4) << 3);
            rowmax[(size_t)b * NN + n] = rmax[r];
            rowsum[(size_t)b * NN + n] = rsum[r];
        }
    }
}

// ---------------- K4: fused energy recompute + softmax + P@V + gating ----------------
__global__ __launch_bounds__(256) void k_attn_out(
    const __bf16* __restrict__ qPack, const __bf16* __restrict__ kPack,
    const __bf16* __restrict__ vhPack, const __bf16* __restrict__ vlPack,
    const float* __restrict__ rowmax, const float* __restrict__ rowsum,
    const float* __restrict__ xh1, const float* __restrict__ xl1,
    const float* __restrict__ g1p, const float* __restrict__ g2p,
    float* __restrict__ outh, float* __restrict__ outl)
{
    __shared__ alignas(64) __bf16 pbuf[8][512];   // per-wave P tile in A-frag layout
    __shared__ alignas(64) float  obuf[8][256];   // per-wave D-tile transpose buffer

    int lane = threadIdx.x & 31;
    int w    = threadIdx.x >> 5;
    int wg = blockIdx.x * 8 + w;
    int cb = wg & 3;
    int nt = (wg >> 2) & (NT - 1);
    int b  = wg >> 10;
    int hi = lane >> 4;
    int l15 = lane & 15;

    v16bf aq = load_frag(qPack + ((((size_t)b * NT + nt) * 32 + lane) << 4));
    float rm[8], ri[8];
#pragma unroll
    for (int r = 0; r < 8; ++r) {
        int n = nt * 16 + r + (hi << 3);
        rm[r] = rowmax[(size_t)b * NN + n];
        ri[r] = 1.0f / rowsum[(size_t)b * NN + n];
    }
    v8f acch[4], accl[4];
#pragma unroll
    for (int i = 0; i < 4; ++i) { acch[i] = vzero8(); accl[i] = vzero8(); }

    for (int mt2 = 0; mt2 < NN / 32; ++mt2) {
#pragma unroll
        for (int t = 0; t < 2; ++t) {
            int mt = mt2 * 2 + t;
            v16bf kf = load_frag(kPack + ((((size_t)b * NT + mt) * 32 + lane) << 4));
            v8f e = wmma_bf16(aq, kf, vzero8());
            int ml = t * 16 + l15;                       // local m in [0,32)
#pragma unroll
            for (int r = 0; r < 8; ++r) {
                float p = __expf(e[r] - rm[r]) * ri[r];
                int nl = r + (hi << 3);                  // local n
                int lpp = nl + (((ml >> 4) & 1) << 4);   // A-frag lane
                int epp = ml & 15;
                pbuf[w][lpp * 16 + epp] = (__bf16)p;
            }
        }
        __builtin_amdgcn_wave_barrier();  // DS is in-order per wave; pin scheduling
        v16bf ap = load_frag(&pbuf[w][lane << 4]);
        size_t vbase = (((size_t)b * (NN / 32) + mt2) * (CC / 16) + cb * 4) * 512 + ((size_t)lane << 4);
#pragma unroll
        for (int ct = 0; ct < 4; ++ct) {
            v16bf bh = load_frag(vhPack + vbase + (size_t)ct * 512);
            acch[ct] = wmma_bf16(ap, bh, acch[ct]);
            v16bf bl = load_frag(vlPack + vbase + (size_t)ct * 512);
            accl[ct] = wmma_bf16(ap, bl, accl[ct]);
        }
        __builtin_amdgcn_wave_barrier();
    }

    float g1 = g1p[0], g2 = g2p[0];
#pragma unroll
    for (int s = 0; s < 2; ++s) {
        const float* xin = s ? xl1 : xh1;
        float* outp      = s ? outl : outh;
        float g          = s ? g2 : g1;
#pragma unroll
        for (int ct = 0; ct < 4; ++ct) {
            v8f a = s ? accl[ct] : acch[ct];
            __builtin_amdgcn_wave_barrier();
#pragma unroll
            for (int r = 0; r < 8; ++r)
                obuf[w][(r + (hi << 3)) * 16 + l15] = a[r] * g;   // [n_local][c_local]
            __builtin_amdgcn_wave_barrier();
            int n = nt * 16 + l15;
#pragma unroll
            for (int r = 0; r < 8; ++r) {
                float t = obuf[w][l15 * 16 + r + (hi << 3)];      // transposed read
                int c = cb * 64 + ct * 16 + r + (hi << 3);
                size_t idx = ((size_t)b * CC + c) * NN + n;       // n-contiguous across lanes
                outp[idx] = xin[idx] * t;
            }
        }
    }
}

extern "C" void kernel_launch(void* const* d_in, const int* in_sizes, int n_in,
                              void* d_out, int out_size, void* d_ws, size_t ws_size,
                              hipStream_t stream) {
    const float* x   = (const float*)d_in[0];
    const float* xh1 = (const float*)d_in[1];
    const float* xl1 = (const float*)d_in[2];
    const float* Wq  = (const float*)d_in[3];
    const float* bq  = (const float*)d_in[4];
    const float* Wk  = (const float*)d_in[5];
    const float* bk  = (const float*)d_in[6];
    const float* Wv  = (const float*)d_in[7];
    const float* bv  = (const float*)d_in[8];
    const float* g1  = (const float*)d_in[9];
    const float* g2  = (const float*)d_in[10];
    float* outh = (float*)d_out;
    float* outl = outh + (size_t)BB * CC * NN;

    char* ws = (char*)d_ws;
    __bf16* qPack  = (__bf16*)(ws);                               // 1 MB
    __bf16* kPack  = (__bf16*)(ws + (size_t)(1u << 20));          // 1 MB
    __bf16* vhPack = (__bf16*)(ws + (size_t)(2u << 20));          // 8 MB
    __bf16* vlPack = (__bf16*)(ws + (size_t)(10u << 20));         // 8 MB
    float*  rowmax = (float*) (ws + (size_t)(18u << 20));         // 64 KB
    float*  rowsum = (float*) (ws + (size_t)(18u << 20) + (1u << 16));

    k_proj_qk<<<(BB * 32 * NN) / 256, 256, 0, stream>>>(x, Wq, bq, Wk, bk, qPack, kPack);
    k_proj_v<<<(BB * NT * 4) / 8, 256, 0, stream>>>(xh1, xl1, Wv, bv, vhPack, vlPack);
    k_stats<<<(BB * NT) / 8, 256, 0, stream>>>(qPack, kPack, rowmax, rowsum);
    k_attn_out<<<(BB * NT * 4) / 8, 256, 0, stream>>>(qPack, kPack, vhPack, vlPack,
                                                      rowmax, rowsum, xh1, xl1, g1, g2,
                                                      outh, outl);
}
